// DebertaV2Attention_71416716198284
// MI455X (gfx1250) — matrix operand order repaired and checked by reading, not compiled, not run on hardware
//
#include <hip/hip_runtime.h>

// ---------------------------------------------------------------------------
// DeBERTa-v2 disentangled attention block for gfx1250 (MI455X), wave32 WMMA.
// All GEMMs use v_wmma_f32_16x16x32_bf16 (bf16 inputs, f32 accum).
// ---------------------------------------------------------------------------

typedef __attribute__((ext_vector_type(16))) __bf16 v16bf;
typedef __attribute__((ext_vector_type(8)))  float  v8f;
typedef __attribute__((ext_vector_type(4)))  int    v4i;

#define S_LEN   1024
#define NHEAD   16
#define HDIM    64
#define DMODEL  1024
#define SPAN    256          // att_span = position_buckets
#define MIDB    128          // BUCKETS/2
#define RSCALE  0.07216878f  // 1/sqrt(64*3)

#if defined(__has_builtin)
#if __has_builtin(__builtin_amdgcn_global_load_async_to_lds_b128)
#define HAVE_ASYNC_LDS 1
#endif
#endif
#ifndef HAVE_ASYNC_LDS
#define HAVE_ASYNC_LDS 0
#endif

#if HAVE_ASYNC_LDS
typedef __attribute__((address_space(1))) v4i* gv4i_p;
typedef __attribute__((address_space(3))) v4i* lv4i_p;
__device__ __forceinline__ void async_copy_b128(const void* g, void* l) {
  __builtin_amdgcn_global_load_async_to_lds_b128((gv4i_p)(void*)g, (lv4i_p)l, 0, 0);
}
#endif

__device__ __forceinline__ v8f wmma_bf16(v16bf a, v16bf b, v8f c) {
  return __builtin_amdgcn_wmma_f32_16x16x32_bf16(false, a, false, b, (short)0, c,
                                                 false, false);
}

// A fragment: 16x32 bf16, element e of lane l holds (m = l&15, k = e+8*half (+8 if e>=8))
__device__ __forceinline__ v16bf load_a_frag(const __bf16* base, int row, int ld,
                                             int k0, int lane) {
  const int m = lane & 15, half = lane >> 4;
  const __bf16* r = base + (size_t)(row + m) * ld + k0;
  v16bf a;
#pragma unroll
  for (int e = 0; e < 16; e += 2) {
    const int k = e + 8 * half + ((e & 8) ? 8 : 0);
    a[e]     = r[k];
    a[e + 1] = r[k + 1];
  }
  return a;
}

// B fragment: 32x16 bf16 with source stored row-major [n][k]:
// lane n = l&15, element e holds k = e + 8*half (+8 if e>=8)
__device__ __forceinline__ v16bf load_b_frag(const __bf16* base, int col, int ld,
                                             int k0, int lane) {
  const int n = lane & 15, half = lane >> 4;
  const __bf16* r = base + (size_t)(col + n) * ld + k0;
  v16bf b;
#pragma unroll
  for (int e = 0; e < 16; ++e) b[e] = r[e + 8 * half + ((e & 8) ? 8 : 0)];
  return b;
}

// DeBERTa log-bucket relative position (build_rel_pos in the reference).
__device__ __forceinline__ int rel_bucket(int rel) {
  int a = (rel < MIDB && rel > -MIDB) ? (MIDB - 1) : abs(rel);
  if (a <= MIDB) return rel;
  const float lp =
      ceilf(logf((float)a / (float)MIDB) / logf(511.0f / (float)MIDB) *
            (float)(MIDB - 1)) + (float)MIDB;
  const int sgn = (rel > 0) - (rel < 0);
  return (int)(lp * (float)sgn);
}

// ---------------------------------------------------------------------------
// Kernel 1: projection GEMM  C[m,n] = sum_k A[m,k]*W[n,k] + bias[n]
// mode 0: out bf16, layout [B,H,S,64]        (QKV projections, M=4096)
// mode 1: out bf16, layout [H,512,64]        (position projections, M=512)
// mode 2: out f32 = C + resid[m,n], row-major [M,1024]  (output projection)
// Block 256 thr = 8 waves; block tile 128x64, wave tile 32x32, BK=32.
// Double-buffered LDS: next tile's global loads issue before the current
// tile's WMMAs and drain underneath them; one barrier per k-iteration.
// ---------------------------------------------------------------------------
__global__ __launch_bounds__(256) void proj_gemm_kernel(
    const float* __restrict__ A, const float* __restrict__ W,
    const float* __restrict__ bias, const float* __restrict__ resid,
    __bf16* __restrict__ outb, float* __restrict__ outf, int M, int K, int mode) {
  __shared__ __bf16 As[2][128 * 32];
  __shared__ __bf16 Bs[2][64 * 32];
  const int tid = threadIdx.x, lane = tid & 31, wave = tid >> 5;
  const int wm = wave >> 1, wn = wave & 1;
  const int m0 = blockIdx.y * 128, n0 = blockIdx.x * 64;

  // per-thread staging coordinates (constant across k-loop)
  int ra[4], ca[4], rb[2], cb[2];
#pragma unroll
  for (int j = 0; j < 4; ++j) {
    const int li = tid + j * 256;            // float4 units, 8 per 32-wide row
    ra[j] = li >> 3; ca[j] = (li & 7) << 2;
  }
#pragma unroll
  for (int j = 0; j < 2; ++j) {
    const int li = tid + j * 256;
    rb[j] = li >> 3; cb[j] = (li & 7) << 2;
  }

  auto load_tile = [&](int k0, float4* fa, float4* fb) {
#pragma unroll
    for (int j = 0; j < 4; ++j)
      fa[j] = *(const float4*)(A + (size_t)(m0 + ra[j]) * K + k0 + ca[j]);
#pragma unroll
    for (int j = 0; j < 2; ++j)
      fb[j] = *(const float4*)(W + (size_t)(n0 + rb[j]) * K + k0 + cb[j]);
  };
  auto store_tile = [&](int buf, const float4* fa, const float4* fb) {
#pragma unroll
    for (int j = 0; j < 4; ++j) {
      __bf16* dst = &As[buf][ra[j] * 32 + ca[j]];
      dst[0] = (__bf16)fa[j].x; dst[1] = (__bf16)fa[j].y;
      dst[2] = (__bf16)fa[j].z; dst[3] = (__bf16)fa[j].w;
    }
#pragma unroll
    for (int j = 0; j < 2; ++j) {
      __bf16* dst = &Bs[buf][rb[j] * 32 + cb[j]];
      dst[0] = (__bf16)fb[j].x; dst[1] = (__bf16)fb[j].y;
      dst[2] = (__bf16)fb[j].z; dst[3] = (__bf16)fb[j].w;
    }
  };

  const int NT = K >> 5;
  float4 fa[4], fb[2];
  load_tile(0, fa, fb);
  store_tile(0, fa, fb);
  __syncthreads();

  v8f acc[2][2] = {};
  for (int kt = 0; kt < NT; ++kt) {
    const int cur = kt & 1;
    const bool more = (kt + 1) < NT;
    if (more) load_tile((kt + 1) << 5, fa, fb);  // overlaps WMMAs below
#pragma unroll
    for (int ti = 0; ti < 2; ++ti) {
      const v16bf af = load_a_frag(As[cur], wm * 32 + ti * 16, 32, 0, lane);
#pragma unroll
      for (int tj = 0; tj < 2; ++tj) {
        const v16bf bf = load_b_frag(Bs[cur], wn * 32 + tj * 16, 32, 0, lane);
        acc[ti][tj] = wmma_bf16(af, bf, acc[ti][tj]);
      }
    }
    if (more) {
      store_tile(1 - cur, fa, fb);
      __syncthreads();
    }
  }

  // epilogue: C layout -> lane n = l&15, VGPR r -> m = r + 8*half
  const int nl = lane & 15, half = lane >> 4;
#pragma unroll
  for (int ti = 0; ti < 2; ++ti)
#pragma unroll
    for (int tj = 0; tj < 2; ++tj)
#pragma unroll
      for (int r = 0; r < 8; ++r) {
        const int m = m0 + wm * 32 + ti * 16 + r + 8 * half;
        const int n = n0 + wn * 32 + tj * 16 + nl;
        float v = acc[ti][tj][r] + bias[n];
        if (mode == 2) {
          const size_t idx = (size_t)m * DMODEL + n;
          outf[idx] = v + resid[idx];
        } else {
          const int h = n >> 6, dd = n & 63;
          size_t idx;
          if (mode == 0)  // [B,H,S,64]
            idx = (((size_t)(m >> 10) * NHEAD + h) * S_LEN + (m & 1023)) * HDIM + dd;
          else            // [H,512,64]
            idx = ((size_t)h * (2 * SPAN) + m) * HDIM + dd;
          outb[idx] = (__bf16)v;
        }
      }
}

// ---------------------------------------------------------------------------
// Kernel 2: position scores  att[bh,m,p] = (X[bh,m,:] . P[h,p,:]) / scale
// X: [BH,S,64] bf16, P: [H,512,64] bf16, att: [BH,S,512] bf16.
// Fragments loaded directly from bf16 row-major global memory (rows = 128B).
// ---------------------------------------------------------------------------
__global__ __launch_bounds__(256) void rel_att_kernel(
    const __bf16* __restrict__ X, const __bf16* __restrict__ P,
    __bf16* __restrict__ att) {
  const int bh = blockIdx.z, h = bh & (NHEAD - 1);
  const int m0 = blockIdx.y * 128, n0 = blockIdx.x * 64;
  const int tid = threadIdx.x, lane = tid & 31, wave = tid >> 5;
  const int wm = wave >> 1, wn = wave & 1;
  const __bf16* Xb = X + (size_t)bh * S_LEN * HDIM;
  const __bf16* Pb = P + (size_t)h * (2 * SPAN) * HDIM;

  v8f acc[2][2] = {};
#pragma unroll
  for (int k0 = 0; k0 < HDIM; k0 += 32)
#pragma unroll
    for (int ti = 0; ti < 2; ++ti) {
      const v16bf af = load_a_frag(Xb, m0 + wm * 32 + ti * 16, HDIM, k0, lane);
#pragma unroll
      for (int tj = 0; tj < 2; ++tj) {
        const v16bf bf = load_b_frag(Pb, n0 + wn * 32 + tj * 16, HDIM, k0, lane);
        acc[ti][tj] = wmma_bf16(af, bf, acc[ti][tj]);
      }
    }

  const int nl = lane & 15, half = lane >> 4;
#pragma unroll
  for (int ti = 0; ti < 2; ++ti)
#pragma unroll
    for (int tj = 0; tj < 2; ++tj)
#pragma unroll
      for (int r = 0; r < 8; ++r) {
        const int m = m0 + wm * 32 + ti * 16 + r + 8 * half;
        const int n = n0 + wn * 32 + tj * 16 + nl;
        att[((size_t)bh * S_LEN + m) * (2 * SPAN) + n] =
            (__bf16)(acc[ti][tj][r] * RSCALE);
      }
}

// ---------------------------------------------------------------------------
// Kernel 3: flash-style attention with disentangled position bias.
// Grid: (S/128 q-tiles, B*H). Block 256 thr = 8 waves; wave owns 16 q rows.
// Q/K tiles move via CDNA5 async load-to-LDS (ASYNCcnt) when available;
// V tile is staged TRANSPOSED ([dd][k]) through VGPRs so the P@V B-fragments
// use the contiguous-k b128 LDS path like all other fragments.
// ---------------------------------------------------------------------------
__global__ __launch_bounds__(256) void attn_kernel(
    const __bf16* __restrict__ qb, const __bf16* __restrict__ kb,
    const __bf16* __restrict__ vb, const __bf16* __restrict__ c2p,
    const __bf16* __restrict__ p2c, const int* __restrict__ mask,
    float* __restrict__ ctx) {
  __shared__ __bf16 Qs[128 * 64];
  __shared__ __bf16 Ks[64 * 64];
  __shared__ __bf16 Vt[64 * 64];   // transposed: Vt[dd][k]
  __shared__ __bf16 Ps[8 * 16 * 64];
  const int bh = blockIdx.y, b = bh >> 4, h = bh & (NHEAD - 1);
  const int q0 = blockIdx.x * 128;
  const int tid = threadIdx.x, lane = tid & 31, wave = tid >> 5;
  const int nl = lane & 15, half = lane >> 4;
  const __bf16* Q = qb + (size_t)bh * S_LEN * HDIM;
  const __bf16* Kp = kb + (size_t)bh * S_LEN * HDIM;
  const __bf16* Vp = vb + (size_t)bh * S_LEN * HDIM;
  const __bf16* c2pb = c2p + (size_t)bh * S_LEN * (2 * SPAN);
  const __bf16* p2cb = p2c + (size_t)bh * S_LEN * (2 * SPAN);

  // stage Q tile (128x64 bf16 = 1024 x b128)
#if HAVE_ASYNC_LDS
#pragma unroll
  for (int j = 0; j < 4; ++j) {
    const int li = tid + j * 256;
    async_copy_b128(&Q[(size_t)q0 * HDIM + li * 8], &Qs[li * 8]);
  }
#else
  {
    uint4 t[4];
#pragma unroll
    for (int j = 0; j < 4; ++j)
      t[j] = *(const uint4*)&Q[(size_t)q0 * HDIM + (tid + j * 256) * 8];
#pragma unroll
    for (int j = 0; j < 4; ++j)
      *(uint4*)&Qs[(tid + j * 256) * 8] = t[j];
  }
#endif

  float rmax[8], rsum[8];
  v8f o[4] = {};
#pragma unroll
  for (int r = 0; r < 8; ++r) { rmax[r] = -1e30f; rsum[r] = 0.0f; }

  for (int kt = 0; kt < S_LEN / 64; ++kt) {
    const int kg0 = kt * 64;
    __syncthreads();  // previous-tile consumers done before Ks/Vt overwrite
    {
      // K tile: async direct to LDS when available
#if HAVE_ASYNC_LDS
#pragma unroll
      for (int j = 0; j < 2; ++j) {
        const int li = tid + j * 256;
        async_copy_b128(&Kp[(size_t)kg0 * HDIM + li * 8], &Ks[li * 8]);
      }
#endif
      uint4 tk[2], tv[2];
#pragma unroll
      for (int j = 0; j < 2; ++j) {
        const int li = tid + j * 256;
#if !HAVE_ASYNC_LDS
        tk[j] = *(const uint4*)&Kp[(size_t)kg0 * HDIM + li * 8];
#endif
        tv[j] = *(const uint4*)&Vp[(size_t)kg0 * HDIM + li * 8];
      }
#pragma unroll
      for (int j = 0; j < 2; ++j) {
        const int li = tid + j * 256;
#if !HAVE_ASYNC_LDS
        *(uint4*)&Ks[li * 8] = tk[j];
#endif
        // V transpose: li covers (krow = li/8, dd0 = (li%8)*8)
        const int krow = li >> 3, dd0 = (li & 7) << 3;
        const __bf16* p = (const __bf16*)&tv[j];
#pragma unroll
        for (int e = 0; e < 8; ++e) Vt[(dd0 + e) * 64 + krow] = p[e];
      }
      (void)tk;
    }
#if HAVE_ASYNC_LDS
    asm volatile("s_wait_asynccnt 0x0" ::: "memory");
#endif
    __syncthreads();

    // --- scores: 16(q) x 64(k) per wave ---
    v8f sc[4] = {};
#pragma unroll
    for (int ks = 0; ks < HDIM; ks += 32) {
      const v16bf af = load_a_frag(Qs, wave * 16, HDIM, ks, lane);
#pragma unroll
      for (int nt = 0; nt < 4; ++nt) {
        const v16bf bf = load_b_frag(Ks, nt * 16, HDIM, ks, lane);
        sc[nt] = wmma_bf16(af, bf, sc[nt]);
      }
    }

    // --- scale + disentangled bias gathers + mask ---
#pragma unroll
    for (int nt = 0; nt < 4; ++nt)
#pragma unroll
      for (int r = 0; r < 8; ++r) {
        const int ig = q0 + wave * 16 + r + 8 * half;
        const int jg = kg0 + nt * 16 + nl;
        const int bkt = rel_bucket(ig - jg);
        const int ci = min(max(bkt + SPAN, 0), 2 * SPAN - 1);
        // p2c index: clip(-bucket(k-q)+span) == clip(bucket(q-k)+span), odd symmetry
        const int pi = ci;
        float v = sc[nt][r] * RSCALE +
                  (float)c2pb[(size_t)ig * (2 * SPAN) + ci] +
                  (float)p2cb[(size_t)jg * (2 * SPAN) + pi];
        if (mask[((size_t)b * S_LEN + ig) * S_LEN + jg] == 0) v = -1e30f;
        sc[nt][r] = v;
      }

    // --- online softmax (row spread over 16 lanes x 4 n-tiles) ---
#pragma unroll
    for (int r = 0; r < 8; ++r) {
      float mx = fmaxf(fmaxf(sc[0][r], sc[1][r]), fmaxf(sc[2][r], sc[3][r]));
#pragma unroll
      for (int off = 1; off < 16; off <<= 1) mx = fmaxf(mx, __shfl_xor(mx, off, 32));
      const float nm = fmaxf(rmax[r], mx);
      const float corr = __expf(rmax[r] - nm);
      rmax[r] = nm;
      float ls = 0.0f;
#pragma unroll
      for (int nt = 0; nt < 4; ++nt) {
        const float p = __expf(sc[nt][r] - nm);
        sc[nt][r] = p;
        ls += p;
      }
#pragma unroll
      for (int off = 1; off < 16; off <<= 1) ls += __shfl_xor(ls, off, 32);
      rsum[r] = rsum[r] * corr + ls;
#pragma unroll
      for (int dt = 0; dt < 4; ++dt) o[dt][r] *= corr;
    }

    // --- stage P (16x64) in per-wave LDS, re-read as WMMA A fragment ---
    __bf16* Pw = &Ps[wave * 16 * 64];
#pragma unroll
    for (int nt = 0; nt < 4; ++nt)
#pragma unroll
      for (int r = 0; r < 8; ++r)
        Pw[(r + 8 * half) * 64 + nt * 16 + nl] = (__bf16)sc[nt][r];
    asm volatile("s_wait_dscnt 0x0" ::: "memory");  // wave-local LDS RAW fence

    // --- O += P @ V  (Vt is [dd][k] so B frags are contiguous-k) ---
#pragma unroll
    for (int ks = 0; ks < 64; ks += 32) {
      const v16bf af = load_a_frag(Pw, 0, 64, ks, lane);
#pragma unroll
      for (int dt = 0; dt < 4; ++dt) {
        const v16bf bf = load_b_frag(Vt, dt * 16, 64, ks, lane);
        o[dt] = wmma_bf16(af, bf, o[dt]);
      }
    }
  }

  // write ctx in [B,S,H,64] (== row-major [4096,1024])
#pragma unroll
  for (int dt = 0; dt < 4; ++dt)
#pragma unroll
    for (int r = 0; r < 8; ++r) {
      const int ig = q0 + wave * 16 + r + 8 * half;
      const int dd = dt * 16 + nl;
      const float denom = rsum[r] > 0.0f ? rsum[r] : 1.0f;
      ctx[(((size_t)b * S_LEN + ig) * NHEAD + h) * HDIM + dd] = o[dt][r] / denom;
    }
}

// ---------------------------------------------------------------------------
// Kernel 4: row LayerNorm (eps 1e-7), one block per row of 1024.
// ---------------------------------------------------------------------------
__global__ __launch_bounds__(256) void ln_kernel(const float* __restrict__ x,
                                                 const float* __restrict__ gamma,
                                                 const float* __restrict__ beta,
                                                 float* __restrict__ out) {
  __shared__ float redS[8], redS2[8];
  __shared__ float mu_s, rstd_s;
  const int row = blockIdx.x, tid = threadIdx.x;
  const float* xr = x + (size_t)row * DMODEL;
  float s = 0.0f, s2 = 0.0f;
  for (int j = tid; j < DMODEL; j += 256) {
    const float v = xr[j];
    s += v; s2 += v * v;
  }
#pragma unroll
  for (int off = 16; off >= 1; off >>= 1) {
    s += __shfl_xor(s, off, 32);
    s2 += __shfl_xor(s2, off, 32);
  }
  if ((tid & 31) == 0) { redS[tid >> 5] = s; redS2[tid >> 5] = s2; }
  __syncthreads();
  if (tid == 0) {
    float S = 0.0f, S2 = 0.0f;
    for (int w = 0; w < 8; ++w) { S += redS[w]; S2 += redS2[w]; }
    const float mu = S / (float)DMODEL;
    mu_s = mu;
    rstd_s = rsqrtf(S2 / (float)DMODEL - mu * mu + 1e-7f);
  }
  __syncthreads();
  const float mu = mu_s, rstd = rstd_s;
  for (int j = tid; j < DMODEL; j += 256)
    out[(size_t)row * DMODEL + j] = (xr[j] - mu) * rstd * gamma[j] + beta[j];
}

// ---------------------------------------------------------------------------
extern "C" void kernel_launch(void* const* d_in, const int* in_sizes, int n_in,
                              void* d_out, int out_size, void* d_ws, size_t ws_size,
                              hipStream_t stream) {
  const float* hidden  = (const float*)d_in[0];
  const int*   mask    = (const int*)d_in[1];
  const float* rel_emb = (const float*)d_in[2];
  const float* Wq = (const float*)d_in[3];
  const float* bq = (const float*)d_in[4];
  const float* Wk = (const float*)d_in[5];
  const float* bk = (const float*)d_in[6];
  const float* Wv = (const float*)d_in[7];
  const float* bv = (const float*)d_in[8];
  const float* Wo = (const float*)d_in[9];
  const float* bo = (const float*)d_in[10];
  const float* gamma = (const float*)d_in[11];
  const float* beta  = (const float*)d_in[12];
  float* out = (float*)d_out;

  const int B = 4, BH = B * NHEAD, M = B * S_LEN;
  char* ws = (char*)d_ws;
  const size_t qkvBytes = (size_t)BH * S_LEN * HDIM * sizeof(__bf16);     // 8 MB
  const size_t posBytes = (size_t)NHEAD * 2 * SPAN * HDIM * sizeof(__bf16);
  const size_t attBytes = (size_t)BH * S_LEN * 2 * SPAN * sizeof(__bf16); // 64 MB
  const size_t f32Bytes = (size_t)M * DMODEL * sizeof(float);             // 16 MB

  __bf16* qb    = (__bf16*)ws;                        ws += qkvBytes;
  __bf16* kbuf  = (__bf16*)ws;                        ws += qkvBytes;
  __bf16* vbuf  = (__bf16*)ws;                        ws += qkvBytes;
  __bf16* posk  = (__bf16*)ws;                        ws += posBytes;
  __bf16* posq  = (__bf16*)ws;                        ws += posBytes;
  __bf16* c2p   = (__bf16*)ws;                        ws += attBytes;
  __bf16* p2c   = (__bf16*)ws;                        ws += attBytes;
  float*  ctx   = (float*)ws;                         ws += f32Bytes;
  float*  preln = (float*)ws;

  const dim3 blk(256);

  // QKV projections -> bf16 [B,H,S,64]
  proj_gemm_kernel<<<dim3(16, 32), blk, 0, stream>>>(hidden, Wq, bq, nullptr, qb,   nullptr, M, DMODEL, 0);
  proj_gemm_kernel<<<dim3(16, 32), blk, 0, stream>>>(hidden, Wk, bk, nullptr, kbuf, nullptr, M, DMODEL, 0);
  proj_gemm_kernel<<<dim3(16, 32), blk, 0, stream>>>(hidden, Wv, bv, nullptr, vbuf, nullptr, M, DMODEL, 0);
  // shared position projections -> bf16 [H,512,64]
  proj_gemm_kernel<<<dim3(16, 4), blk, 0, stream>>>(rel_emb, Wk, bk, nullptr, posk, nullptr, 2 * SPAN, DMODEL, 1);
  proj_gemm_kernel<<<dim3(16, 4), blk, 0, stream>>>(rel_emb, Wq, bq, nullptr, posq, nullptr, 2 * SPAN, DMODEL, 1);
  // c2p_att = Q @ posK^T / scale ; p2c_att = K @ posQ^T / scale
  rel_att_kernel<<<dim3(8, 8, BH), blk, 0, stream>>>(qb,   posk, c2p);
  rel_att_kernel<<<dim3(8, 8, BH), blk, 0, stream>>>(kbuf, posq, p2c);
  // fused disentangled attention + online softmax -> ctx [B,S,H*d] f32
  attn_kernel<<<dim3(S_LEN / 128, BH), blk, 0, stream>>>(qb, kbuf, vbuf, c2p, p2c, mask, ctx);
  // output projection + bias + residual -> preln
  proj_gemm_kernel<<<dim3(16, 32), blk, 0, stream>>>(ctx, Wo, bo, hidden, nullptr, preln, M, DMODEL, 2);
  // LayerNorm -> out
  ln_kernel<<<M, blk, 0, stream>>>(preln, gamma, beta, out);
}
